// EncoderDecoderLSTM_65300682768673
// MI455X (gfx1250) — matrix operand order repaired and checked by reading, not compiled, not run on hardware
//
#include <hip/hip_runtime.h>

// ---------------------------------------------------------------------------
// EncoderDecoder LSTM for MI455X (gfx1250, wave32, WMMA f16 -> f32)
//   B=8192, T=336, H=64, HORIZON=168
// Each wave owns 16 batch rows; h state bounces through a per-wave LDS buffer
// in the WMMA A-fragment-friendly layout; c state lives in f32 registers;
// weights are pre-packed (f16, B-fragment layout) into d_ws and staged to LDS.
// ---------------------------------------------------------------------------

typedef _Float16 h2  __attribute__((ext_vector_type(2)));
typedef _Float16 h8  __attribute__((ext_vector_type(8)));
typedef _Float16 v16h __attribute__((ext_vector_type(16)));
typedef float    v8f  __attribute__((ext_vector_type(8)));

#define HSTR   88            // h-buffer row stride in halves (bank-conflict free, 16B aligned)
#define T_ENC  336
#define T_DEC  168
#define WS_F32_DW (7 * 8192) // 7 packed 32KB matrices, then f32 constants

__device__ __forceinline__ float fast_sig(float x)  { return 1.0f / (1.0f + __expf(-x)); }
__device__ __forceinline__ float fast_tanh(float x) { return 1.0f - 2.0f / (1.0f + __expf(2.0f * x)); }

__device__ __forceinline__ v16h ld_frag2(const _Float16* p0, const _Float16* p1) {
    h8 a = *(const h8*)p0;
    h8 b = *(const h8*)p1;
    return __builtin_shufflevector(a, b, 0,1,2,3,4,5,6,7,8,9,10,11,12,13,14,15);
}

// A-matrix fragments (16x32 f16) for K=0..31 and K=32..63 of a 16x64 h tile.
__device__ __forceinline__ void ld_a(const _Float16* hbuf, int lane, v16h& f0, v16h& f1) {
    const _Float16* r = hbuf + (lane & 15) * HSTR + ((lane >> 4) * 8);
    f0 = ld_frag2(r,      r + 16);
    f1 = ld_frag2(r + 32, r + 48);
}

__device__ __forceinline__ v8f wmma_f16(v16h a, v16h b, v8f c) {
    return __builtin_amdgcn_wmma_f32_16x16x32_f16(false, a, false, b, (short)0, c, false, false);
}

// One LSTM cell for a 16-row tile. Gate preacts G[16x256] = bias (+ x-term) +
// A_ih @ W_ih^T + A_hh @ W_hh^T, then activations, c update, h -> LDS (f16).
// MODE 0: scalar-x input (encoder L0) + hh matmul
// MODE 1: ih matmul + hh matmul
// MODE 2: hh matmul only (decoder L0, step 0: zero input)
template <int MODE>
__device__ __forceinline__ void lstm_gates(
    const _Float16* wih, const _Float16* whh, const float* bias,
    const float* xw, const float* xv,                 // MODE 0 only
    v16h aih0, v16h aih1, v16h ahh0, v16h ahh1,
    v8f cst[4], _Float16* hout, int lane)
{
    const int nl = lane & 15;
    const int hi = lane >> 4;
#pragma unroll
    for (int j = 0; j < 4; ++j) {                     // i/f/g/o column quadruple
        v8f acc[4];
#pragma unroll
        for (int q = 0; q < 4; ++q) {
            const int nt = j + 4 * q;                 // n-tile (gate columns 16nt..16nt+15)
            const float b = bias[nt * 16 + nl];
#pragma unroll
            for (int v = 0; v < 8; ++v) acc[q][v] = b;
            if (MODE == 0) {
                const float w = xw[nt * 16 + nl];
#pragma unroll
                for (int v = 0; v < 8; ++v) acc[q][v] += w * xv[v];
            }
            if (MODE == 1) {
                const _Float16* p0 = wih + (( 0 + nt) * 32 + lane) * 16;
                acc[q] = wmma_f16(aih0, ld_frag2(p0, p0 + 8), acc[q]);
                const _Float16* p1 = wih + ((16 + nt) * 32 + lane) * 16;
                acc[q] = wmma_f16(aih1, ld_frag2(p1, p1 + 8), acc[q]);
            }
            const _Float16* q0 = whh + (( 0 + nt) * 32 + lane) * 16;
            acc[q] = wmma_f16(ahh0, ld_frag2(q0, q0 + 8), acc[q]);
            const _Float16* q1 = whh + ((16 + nt) * 32 + lane) * 16;
            acc[q] = wmma_f16(ahh1, ld_frag2(q1, q1 + 8), acc[q]);
        }
#pragma unroll
        for (int v = 0; v < 8; ++v) {
            const float ig = fast_sig(acc[0][v]);
            const float fg = fast_sig(acc[1][v]);
            const float gg = fast_tanh(acc[2][v]);
            const float og = fast_sig(acc[3][v]);
            const float cn = fg * cst[j][v] + ig * gg;
            cst[j][v] = cn;
            const float hn = og * fast_tanh(cn);
            hout[(v + 8 * hi) * HSTR + (j * 16 + nl)] = (_Float16)hn;
        }
    }
}

// ---------------------------------------------------------------------------
// Pack kernel: f32 weights -> f16 in exact B-fragment layout, plus fused
// biases / Wih0 column / FC params, all into d_ws.
// B-fragment element map: n = lane%16, k = (lane/16)*8 + (v/4)*16 + 2(v%4) + h.
// ---------------------------------------------------------------------------
__global__ void lstm_pack(
    const float* eW0ih, const float* eW0hh, const float* eb0i, const float* eb0h,
    const float* eW1ih, const float* eW1hh, const float* eb1i, const float* eb1h,
    const float* dW0ih, const float* dW0hh, const float* db0i, const float* db0h,
    const float* dW1ih, const float* dW1hh, const float* db1i, const float* db1h,
    const float* fcW, const float* fcB, float* wsf)
{
    const int id = blockIdx.x * 256 + threadIdx.x;
    if (id < 7 * 8192) {
        const int mat = id >> 13;
        const int r   = id & 8191;
        const int v   = r & 7;
        const int ln  = (r >> 3) & 31;
        const int f   = r >> 8;
        const int j   = f & 15;
        const int kt  = f >> 4;
        const int grow = j * 16 + (ln & 15);
        const int k    = kt * 32 + (ln >> 4) * 8 + (v >> 2) * 16 + 2 * (v & 3);
        const float* W;
        switch (mat) {
            case 0: W = eW0hh; break;
            case 1: W = eW1ih; break;
            case 2: W = eW1hh; break;
            case 3: W = dW0ih; break;
            case 4: W = dW0hh; break;
            case 5: W = dW1ih; break;
            default: W = dW1hh; break;
        }
        h2 p;
        p.x = (_Float16)W[grow * 64 + k];
        p.y = (_Float16)W[grow * 64 + k + 1];
        ((h2*)wsf)[id] = p;
    } else {
        const int i = id - 7 * 8192;
        if      (i <  256) wsf[WS_F32_DW + i] = eb0i[i]        + eb0h[i];
        else if (i <  512) wsf[WS_F32_DW + i] = eb1i[i -  256] + eb1h[i -  256];
        else if (i <  768) wsf[WS_F32_DW + i] = db0i[i -  512] + db0h[i -  512];
        else if (i < 1024) wsf[WS_F32_DW + i] = db1i[i -  768] + db1h[i -  768];
        else if (i < 1280) wsf[WS_F32_DW + i] = eW0ih[i - 1024];
        else if (i < 1344) wsf[WS_F32_DW + i] = fcW[i - 1280];
        else if (i == 1344) wsf[WS_F32_DW + i] = fcB[0];
    }
}

// ---------------------------------------------------------------------------
// Main persistent kernel: 64 blocks x 256 threads (8 waves), 16 rows / wave.
// Dynamic LDS: [0,128K) weight slots, [128K,..) biases/consts, then per-wave
// h0/h1 bounce buffers.
// ---------------------------------------------------------------------------
__global__ __launch_bounds__(256, 1)
void lstm_main(const float* __restrict__ x, const float* __restrict__ wsf,
               float* __restrict__ out)
{
    extern __shared__ char smem[];
    _Float16* wslot  = (_Float16*)smem;                 // 4 slots x 32KB
    float*    consts = (float*)(smem + 131072);         // 1345 f32
    const float* encb0 = consts;
    const float* encb1 = consts + 256;
    const float* decb0 = consts + 512;
    const float* decb1 = consts + 768;
    const float* wih0c = consts + 1024;
    const float* fcw   = consts + 1280;

    const int tid  = threadIdx.x;
    const int lane = tid & 31;
    const int wave = tid >> 5;
    const int hi   = lane >> 4;

    _Float16* hbuf0 = (_Float16*)(smem + 136704 + wave * 5632);
    _Float16* hbuf1 = hbuf0 + 16 * HSTR;

    // Stage f32 constants and encoder weights (3 matrices -> slots 0..2).
    for (int i = tid; i < 1345; i += 256) consts[i] = wsf[WS_F32_DW + i];
    {
        const float4* src = (const float4*)wsf;
        float4* dst = (float4*)smem;
        for (int i = tid; i < 6144; i += 256) dst[i] = src[i];
    }
    {   // zero this wave's h buffers
        uint32_t* hz = (uint32_t*)hbuf0;
        for (int i = lane; i < 1408; i += 32) hz[i] = 0u;
    }
    __syncthreads();

    const int rowbase = (blockIdx.x * 8 + wave) * 16;

    v8f c0[4], c1[4];
#pragma unroll
    for (int j = 0; j < 4; ++j)
#pragma unroll
        for (int v = 0; v < 8; ++v) { c0[j][v] = 0.0f; c1[j][v] = 0.0f; }

    v16h z;
#pragma unroll
    for (int e = 0; e < 16; ++e) z[e] = (_Float16)0.0f;

    // ------------------------- Encoder (336 steps) -------------------------
    {
        const _Float16* w_e0hh = wslot;
        const _Float16* w_e1ih = wslot + 16384;
        const _Float16* w_e1hh = wslot + 32768;

        for (int t = 0; t < T_ENC; ++t) {
            float myx = 0.0f;
            if (lane < 16) myx = x[(rowbase + lane) * T_ENC + t];
            float xv[8];
#pragma unroll
            for (int v = 0; v < 8; ++v) xv[v] = __shfl(myx, v + 8 * hi, 32);

            v16h a0, a1;
            ld_a(hbuf0, lane, a0, a1);                       // h0 prev
            lstm_gates<0>(nullptr, w_e0hh, encb0, wih0c, xv,
                          z, z, a0, a1, c0, hbuf0, lane);

            v16h b0, b1, d0, d1;
            ld_a(hbuf0, lane, b0, b1);                       // h0 new
            ld_a(hbuf1, lane, d0, d1);                       // h1 prev
            lstm_gates<1>(w_e1ih, w_e1hh, encb1, nullptr, nullptr,
                          b0, b1, d0, d1, c1, hbuf1, lane);
        }
    }

    // Swap in decoder weights (4 matrices -> slots 0..3).
    __syncthreads();
    {
        const float4* src = (const float4*)wsf;
        float4* dst = (float4*)smem;
        for (int i = tid; i < 8192; i += 256) dst[i] = src[6144 + i];
    }
    __syncthreads();

    // ------------------------- Decoder (168 steps) -------------------------
    {
        const _Float16* w_d0ih = wslot;
        const _Float16* w_d0hh = wslot + 16384;
        const _Float16* w_d1ih = wslot + 32768;
        const _Float16* w_d1hh = wslot + 49152;
        const float fcb = consts[1344];

        for (int t = 0; t < T_DEC; ++t) {
            v16h x0, x1, a0, a1;
            ld_a(hbuf1, lane, x0, x1);                       // xin = h1 prev
            ld_a(hbuf0, lane, a0, a1);                       // h0 prev
            if (t == 0)
                lstm_gates<2>(nullptr, w_d0hh, decb0, nullptr, nullptr,
                              z, z, a0, a1, c0, hbuf0, lane);
            else
                lstm_gates<1>(w_d0ih, w_d0hh, decb0, nullptr, nullptr,
                              x0, x1, a0, a1, c0, hbuf0, lane);

            v16h n0, n1;
            ld_a(hbuf0, lane, n0, n1);                       // h0 new
            lstm_gates<1>(w_d1ih, w_d1hh, decb1, nullptr, nullptr,
                          n0, n1, x0, x1, c1, hbuf1, lane);

            if (lane < 16) {                                 // FC head
                float s = fcb;
                const _Float16* hr = hbuf1 + lane * HSTR;
#pragma unroll
                for (int u = 0; u < 64; ++u) s += (float)hr[u] * fcw[u];
                out[(rowbase + lane) * T_DEC + t] = s;
            }
        }
    }
}

extern "C" void kernel_launch(void* const* d_in, const int* in_sizes, int n_in,
                              void* d_out, int out_size, void* d_ws, size_t ws_size,
                              hipStream_t stream)
{
    const float* x     = (const float*)d_in[0];
    const float* eW0ih = (const float*)d_in[1];
    const float* eW0hh = (const float*)d_in[2];
    const float* eb0i  = (const float*)d_in[3];
    const float* eb0h  = (const float*)d_in[4];
    const float* eW1ih = (const float*)d_in[5];
    const float* eW1hh = (const float*)d_in[6];
    const float* eb1i  = (const float*)d_in[7];
    const float* eb1h  = (const float*)d_in[8];
    const float* dW0ih = (const float*)d_in[9];
    const float* dW0hh = (const float*)d_in[10];
    const float* db0i  = (const float*)d_in[11];
    const float* db0h  = (const float*)d_in[12];
    const float* dW1ih = (const float*)d_in[13];
    const float* dW1hh = (const float*)d_in[14];
    const float* db1i  = (const float*)d_in[15];
    const float* db1h  = (const float*)d_in[16];
    const float* fcW   = (const float*)d_in[17];
    const float* fcB   = (const float*)d_in[18];
    float* wsf = (float*)d_ws;
    float* out = (float*)d_out;

    const int packThreads = 7 * 8192 + 1345;
    const int packBlocks  = (packThreads + 255) / 256;
    lstm_pack<<<packBlocks, 256, 0, stream>>>(
        eW0ih, eW0hh, eb0i, eb0h, eW1ih, eW1hh, eb1i, eb1h,
        dW0ih, dW0hh, db0i, db0h, dW1ih, dW1hh, db1i, db1h,
        fcW, fcB, wsf);

    const size_t smemBytes = 181760;  // 128K weights + 5.4K consts + 44K h-buffers
    lstm_main<<<64, 256, smemBytes, stream>>>(x, wsf, out);
}
